// QKVAttention_6021544149580
// MI455X (gfx1250) — matrix operand order repaired and checked by reading, not compiled, not run on hardware
//
#include <hip/hip_runtime.h>
#include <hip/hip_bf16.h>

typedef _Float16 h16;
typedef __attribute__((ext_vector_type(16))) _Float16 v16h;
typedef __attribute__((ext_vector_type(8)))  float    v8f;
typedef __attribute__((ext_vector_type(4)))  unsigned int v4u;

// Problem constants (fixed by the reference setup_inputs()).
#define TT      5120   // token dim of qkv
#define WIDTH   1536   // qkv width (3 * 8 heads * 64 ch)
#define CH      64     // channels per head
#define WW      320    // attention window
#define VLEN    4096
#define NFRAMES 16

#define KS_STR  72     // halves per Ks row  (144 B, 16B aligned)
#define VT_STR  336    // halves per Vt row  (672 B, 16B aligned)
#define PS_STR  40     // halves per Ps row  (80 B, 16B aligned)
#define OS_STR  20     // floats per Os row  (80 B, 16B aligned, conflict-free)
#define WS_BYTES 5120  // per-wave scratch: max(16*PS_STR*2, 64*OS_STR*4)

static __device__ __forceinline__ v16h ld_b16(const h16* p) {
  // B-matrix fragment: 16 contiguous halves starting at p (16B aligned).
  v16h r;
  ((v4u*)&r)[0] = *(const v4u*)(p);
  ((v4u*)&r)[1] = *(const v4u*)(p + 8);
  return r;
}

__global__ __launch_bounds__(256)
void windowed_attn_kernel(const float* __restrict__ qkv,
                          const int*   __restrict__ vidx,
                          const int*   __restrict__ aidx,
                          float*       __restrict__ out) {
  extern __shared__ char smem[];
  h16*  Ks     = (h16*)smem;                                   // [WW][KS_STR]
  h16*  Vt     = (h16*)(smem + WW * KS_STR * 2);               // [CH][VT_STR]
  char* wsbase = smem + WW * KS_STR * 2 + CH * VT_STR * 2;     // 8 * WS_BYTES

  const int bx = blockIdx.x;
  const int f  = bx & 15;
  const int hh = (bx >> 4) & 7;
  const int b  = bx >> 7;
  const bool audio = (blockIdx.y != 0);

  const int tq      = audio ? 64 : 256;                 // queries per frame
  const int seg_len = audio ? 1024 : 4096;              // output token length
  const size_t out_seg = audio ? (size_t)4 * 512 * 4096 : 0;
  const int qt0 = audio ? (VLEN + f * 64) : (f * 256);  // q column offset in T
  const int* idxrow = audio ? (aidx + (size_t)(f * 64) * WW)
                            : (vidx + (size_t)(f * 256) * WW);

  const float* Qb = qkv + (size_t)(b * WIDTH + hh * CH) * TT;
  const float* Kb = qkv + (size_t)(b * WIDTH + 512 + hh * CH) * TT;
  const float* Vb = qkv + (size_t)(b * WIDTH + 1024 + hh * CH) * TT;

  const int tid = threadIdx.x;

  // ---- Phase 1: gather K/V window into LDS as f16 (K row-major, V transposed)
  for (int e = tid; e < WW * CH; e += 256) {
    const int ch = e / WW;
    const int w  = e - ch * WW;
    const int col = idxrow[w];
    Ks[w * KS_STR + ch]  = (h16)Kb[(size_t)ch * TT + col];
    Vt[ch * VT_STR + w]  = (h16)Vb[(size_t)ch * TT + col];
  }
  __syncthreads();

  const int wave = tid >> 5;
  const int lane = tid & 31;
  const int ml = lane & 15;   // row (A) or column (B/C) index within tile
  const int kh = lane >> 4;   // lane-half: selects K-range / M-range per ISA layout

  h16*   Ps = (h16*)(wsbase + wave * WS_BYTES);   // [16][PS_STR] P bounce
  float* Os = (float*)(wsbase + wave * WS_BYTES); // [64][OS_STR] out bounce (aliased; disjoint phases)

  const int nstrips = tq >> 4;
  for (int strip = wave; strip < nstrips; strip += 8) {
    const int tcol = qt0 + strip * 16 + ml;

    // ---- Q A-fragments straight from global in k = 16*kh + s order; fold scale^2=0.125.
    v16h aq0, aq1;
    #pragma unroll
    for (int j = 0; j < 8; ++j) {
      aq0[j]     = (h16)(Qb[(size_t)(     kh * 16 + j) * TT + tcol] * 0.125f);
      aq0[j + 8] = (h16)(Qb[(size_t)( 8 + kh * 16 + j) * TT + tcol] * 0.125f);
      aq1[j]     = (h16)(Qb[(size_t)(32 + kh * 16 + j) * TT + tcol] * 0.125f);
      aq1[j + 8] = (h16)(Qb[(size_t)(40 + kh * 16 + j) * TT + tcol] * 0.125f);
    }

    // ---- QK^T: 20 key tiles of 16, K-dim 64 = 2 WMMAs per tile.
    v8f sv[20];
    #pragma unroll
    for (int kt = 0; kt < 20; ++kt) {
      const h16* krow = Ks + (kt * 16 + ml) * KS_STR + kh * 16;
      v16h bk0 = ld_b16(krow);        // ch 0..31  (this lane-half's 16)
      v16h bk1 = ld_b16(krow + 32);   // ch 32..63
      v8f acc = {};
      acc = __builtin_amdgcn_wmma_f32_16x16x32_f16(false, aq0, false, bk0,
                                                   (short)0, acc, false, false);
      acc = __builtin_amdgcn_wmma_f32_16x16x32_f16(false, aq1, false, bk1,
                                                   (short)0, acc, false, false);
      sv[kt] = acc;
    }

    // ---- Row softmax over 320 keys. Row m lives in vgpr m%8, lane-half m/8;
    //      reduce across the 16 key-lanes with width-16 xor shuffles.
    float mx[8], sm[8];
    #pragma unroll
    for (int r = 0; r < 8; ++r) mx[r] = -1e30f;
    #pragma unroll
    for (int kt = 0; kt < 20; ++kt) {
      #pragma unroll
      for (int r = 0; r < 8; ++r) mx[r] = fmaxf(mx[r], sv[kt][r]);
    }
    #pragma unroll
    for (int r = 0; r < 8; ++r) {
      #pragma unroll
      for (int d = 1; d < 16; d <<= 1) mx[r] = fmaxf(mx[r], __shfl_xor(mx[r], d, 16));
      sm[r] = 0.0f;
    }
    #pragma unroll
    for (int kt = 0; kt < 20; ++kt) {
      #pragma unroll
      for (int r = 0; r < 8; ++r) {
        const float e = __expf(sv[kt][r] - mx[r]);
        sv[kt][r] = e;
        sm[r] += e;
      }
    }
    #pragma unroll
    for (int r = 0; r < 8; ++r) {
      #pragma unroll
      for (int d = 1; d < 16; d <<= 1) sm[r] += __shfl_xor(sm[r], d, 16);
      sm[r] = 1.0f / sm[r];
    }

    // ---- O = P * V : 10 key-chunks of 32, 4 ch tiles accumulated.
    v8f o0 = {}, o1 = {}, o2 = {}, o3 = {};
    #pragma unroll
    for (int ck = 0; ck < 10; ++ck) {
      // C/D layout -> A layout via per-wave LDS bounce (same-wave DS is in-order).
      #pragma unroll
      for (int t2 = 0; t2 < 2; ++t2) {
        const int kt = ck * 2 + t2;
        #pragma unroll
        for (int r = 0; r < 8; ++r)
          Ps[(r + 8 * kh) * PS_STR + t2 * 16 + ml] = (h16)(sv[kt][r] * sm[r]);
      }
      __builtin_amdgcn_wave_barrier();
      v16h pa;
      {
        const h16* prow = Ps + ml * PS_STR + kh * 16;
        ((v4u*)&pa)[0] = *(const v4u*)(prow);        // k_local kh*16+0..7
        ((v4u*)&pa)[1] = *(const v4u*)(prow + 8);    // k_local kh*16+8..15
      }
      __builtin_amdgcn_wave_barrier();
      const int kb = ck * 32 + kh * 16;
      o0 = __builtin_amdgcn_wmma_f32_16x16x32_f16(false, pa, false,
             ld_b16(Vt + ( 0 + ml) * VT_STR + kb), (short)0, o0, false, false);
      o1 = __builtin_amdgcn_wmma_f32_16x16x32_f16(false, pa, false,
             ld_b16(Vt + (16 + ml) * VT_STR + kb), (short)0, o1, false, false);
      o2 = __builtin_amdgcn_wmma_f32_16x16x32_f16(false, pa, false,
             ld_b16(Vt + (32 + ml) * VT_STR + kb), (short)0, o2, false, false);
      o3 = __builtin_amdgcn_wmma_f32_16x16x32_f16(false, pa, false,
             ld_b16(Vt + (48 + ml) * VT_STR + kb), (short)0, o3, false, false);
    }

    // ---- Bounce output tile [64 ch][16 t] through LDS, store coalesced float4s.
    #pragma unroll
    for (int r = 0; r < 8; ++r) {
      const int m = r + 8 * kh;
      Os[( 0 + ml) * OS_STR + m] = o0[r];
      Os[(16 + ml) * OS_STR + m] = o1[r];
      Os[(32 + ml) * OS_STR + m] = o2[r];
      Os[(48 + ml) * OS_STR + m] = o3[r];
    }
    __builtin_amdgcn_wave_barrier();

    const int tbase = (audio ? f * 64 : f * 256) + strip * 16;
    float* outp = out + out_seg + (size_t)(b * 512 + hh * CH) * seg_len + tbase;
    #pragma unroll
    for (int p = 0; p < 8; ++p) {
      const int c  = p * 8 + (lane >> 2);
      const int m0 = (lane & 3) * 4;
      const float4 val = *(const float4*)(Os + c * OS_STR + m0);
      *(float4*)(outp + (size_t)c * seg_len + m0) = val;
    }
    __builtin_amdgcn_wave_barrier();
  }
}

extern "C" void kernel_launch(void* const* d_in, const int* in_sizes, int n_in,
                              void* d_out, int out_size, void* d_ws, size_t ws_size,
                              hipStream_t stream) {
  (void)in_sizes; (void)n_in; (void)out_size; (void)d_ws; (void)ws_size;
  const float* qkv  = (const float*)d_in[0];
  const int*   vidx = (const int*)d_in[1];
  const int*   aidx = (const int*)d_in[2];
  float* out = (float*)d_out;

  const size_t lds_bytes = (size_t)WW * KS_STR * 2   // Ks
                         + (size_t)CH * VT_STR * 2   // Vt
                         + (size_t)8 * WS_BYTES;     // per-wave scratch
  dim3 grid(4 * 8 * NFRAMES, 2);  // (b,h,f) x {video,audio}
  dim3 block(256);
  windowed_attn_kernel<<<grid, block, lds_bytes, stream>>>(qkv, vidx, aidx, out);
}
